// TemporalGraphAttention_25005299597835
// MI455X (gfx1250) — compile-verified, matrix operand order checked by reading, not run on hardware
//
#include <hip/hip_runtime.h>
#include <hip/hip_bf16.h>

typedef __attribute__((ext_vector_type(16))) __bf16 v16bf;
typedef __attribute__((ext_vector_type(8)))  float  v8f;

// ---------------------------------------------------------------------------
// Zero-fill (harness poisons d_out / d_ws with 0xAA; we accumulate with atomics)
// ---------------------------------------------------------------------------
__global__ void k_fill_zero(float* __restrict__ p, int n) {
    int i = blockIdx.x * blockDim.x + threadIdx.x;
    if (i < n) p[i] = 0.0f;
}

// ---------------------------------------------------------------------------
// Pack W [H=8][F=128][O=16] (f32, row-major) into bf16 B-fragments laid out
// exactly as v_wmma_f32_16x16x32_bf16 wants them, per lane:
//   entry i = ((t*4 + kf)*32 + lane)*16 + j
//   K(j,lane) = kf*32 + (j<8 ? j : j+8) + (lane>=16 ? 8 : 0),  N = lane&15
// ---------------------------------------------------------------------------
__global__ void k_pack_W(const float* __restrict__ W, __bf16* __restrict__ wp) {
    for (int i = threadIdx.x; i < 8 * 4 * 32 * 16; i += blockDim.x) {
        int j    = i & 15;
        int lane = (i >> 4) & 31;
        int kf   = (i >> 9) & 3;
        int t    = (i >> 11) & 7;
        int k    = kf * 32 + (j < 8 ? j : j + 8) + ((lane >= 16) ? 8 : 0);
        int n    = lane & 15;
        wp[i] = (__bf16)W[(t * 128 + k) * 16 + n];   // W[t][k][n]
    }
}

// ---------------------------------------------------------------------------
// GEMM: h[N,128] = x[N,128] @ Wflat[128,128] via WMMA bf16 -> f32.
// One wave = one 16-row strip x all 8 head-tiles (8 x v8f accumulators).
// ---------------------------------------------------------------------------
__global__ void __launch_bounds__(256)
k_gemm_wmma(const float* __restrict__ x, const __bf16* __restrict__ wpack,
            float* __restrict__ h, int nNodes) {
    const int lane     = threadIdx.x & 31;
    const int wave     = blockIdx.x * 8 + (threadIdx.x >> 5);
    const int m0       = wave * 16;
    if (m0 >= nNodes) return;                 // uniform per-wave guard (EXEC stays all-1s)
    const int laneHigh = lane >> 4;
    int row = m0 + (lane & 15);
    if (row >= nNodes) row = nNodes - 1;      // clamp (N=100000 is a multiple of 16 anyway)
    const float* xrow = x + (long)row * 128;

    v8f acc[8] = {};
    const v16bf* wfrag = (const v16bf*)wpack;

    #pragma unroll
    for (int kf = 0; kf < 4; ++kf) {
        // Build bf16 A fragment: two contiguous 8-float chunks of this lane's row.
        const int c0 = kf * 32 + laneHigh * 8;
        float4 f0 = *(const float4*)(xrow + c0);
        float4 f1 = *(const float4*)(xrow + c0 + 4);
        float4 f2 = *(const float4*)(xrow + c0 + 16);
        float4 f3 = *(const float4*)(xrow + c0 + 20);
        v16bf afrag;
        afrag[0]  = (__bf16)f0.x; afrag[1]  = (__bf16)f0.y; afrag[2]  = (__bf16)f0.z; afrag[3]  = (__bf16)f0.w;
        afrag[4]  = (__bf16)f1.x; afrag[5]  = (__bf16)f1.y; afrag[6]  = (__bf16)f1.z; afrag[7]  = (__bf16)f1.w;
        afrag[8]  = (__bf16)f2.x; afrag[9]  = (__bf16)f2.y; afrag[10] = (__bf16)f2.z; afrag[11] = (__bf16)f2.w;
        afrag[12] = (__bf16)f3.x; afrag[13] = (__bf16)f3.y; afrag[14] = (__bf16)f3.z; afrag[15] = (__bf16)f3.w;

        #pragma unroll
        for (int t = 0; t < 8; ++t) {
            v16bf bfrag = wfrag[(t * 4 + kf) * 32 + lane];   // coalesced 32B/lane
            acc[t] = __builtin_amdgcn_wmma_f32_16x16x32_bf16(
                false, afrag, false, bfrag, (short)0, acc[t], false, false);
        }
    }

    // D layout: lanes 0-15: N=lane, M=r; lanes 16-31: N=lane-16, M=r+8.
    // For fixed r, half-wave writes one contiguous 64B row segment.
    #pragma unroll
    for (int t = 0; t < 8; ++t) {
        #pragma unroll
        for (int r = 0; r < 8; ++r) {
            int rr = m0 + r + laneHigh * 8;
            if (rr < nNodes)
                h[(long)rr * 128 + t * 16 + (lane & 15)] = acc[t][r];
        }
    }
}

// ---------------------------------------------------------------------------
// Per-node attention partials: alpha[n*16+h]   = <h[n,h,:], a[h,0:16]>
//                              alpha[n*16+8+h] = <h[n,h,:], a[h,16:32]>
// ---------------------------------------------------------------------------
__global__ void k_alpha(const float* __restrict__ h, const float* __restrict__ a,
                        float* __restrict__ alpha, int nNodes) {
    int i = blockIdx.x * blockDim.x + threadIdx.x;   // node*8 + head
    if (i >= nNodes * 8) return;
    int node = i >> 3, hd = i & 7;
    const float* hp = h + (long)node * 128 + hd * 16;
    const float* ap = a + hd * 32;
    float s = 0.f, d = 0.f;
    #pragma unroll
    for (int o = 0; o < 16; ++o) { float v = hp[o]; s += v * ap[o]; d += v * ap[16 + o]; }
    alpha[node * 16 + hd]     = s;
    alpha[node * 16 + 8 + hd] = d;
}

// ---------------------------------------------------------------------------
// Edge pass 1: softmax denominators (one thread per edge, 8 heads, L2 atomics)
// ---------------------------------------------------------------------------
__global__ void k_edge_denom(const int* __restrict__ ei, const int* __restrict__ ts,
                             const float* __restrict__ alpha, const float* __restrict__ twp,
                             float* __restrict__ denom, int E) {
    int e = blockIdx.x * blockDim.x + threadIdx.x;
    if (e >= E) return;
    int s = ei[e], d = ei[E + e];
    float dt    = fabsf((float)(ts[s] - ts[d]));
    float decay = __expf(-twp[0] * dt);
    #pragma unroll
    for (int hd = 0; hd < 8; ++hd) {
        float eh = (alpha[s * 16 + hd] + alpha[d * 16 + 8 + hd]) * decay;
        eh = eh > 0.f ? eh : 0.2f * eh;               // LeakyReLU(0.2)
        atomicAdd(&denom[d * 8 + hd], __expf(eh));
    }
}

// ---------------------------------------------------------------------------
// Edge pass 2: one wave per edge; lane owns 4 of the 128 output features.
// out[dst] += att(head) * h[src]  (scatter-add, L2 atomics)
// ---------------------------------------------------------------------------
__global__ void k_edge_scatter(const int* __restrict__ ei, const int* __restrict__ ts,
                               const float* __restrict__ alpha, const float* __restrict__ twp,
                               const float* __restrict__ denom, const float* __restrict__ h,
                               float* __restrict__ out, int E) {
    int tid  = blockIdx.x * blockDim.x + threadIdx.x;
    int e    = tid >> 5;
    if (e >= E) return;
    int lane = tid & 31;
    int hd   = lane >> 2;
    int o4   = (lane & 3) * 4;
    int s = ei[e], d = ei[E + e];
    float dt    = fabsf((float)(ts[s] - ts[d]));
    float decay = __expf(-twp[0] * dt);
    float eh = (alpha[s * 16 + hd] + alpha[d * 16 + 8 + hd]) * decay;
    eh = eh > 0.f ? eh : 0.2f * eh;
    float att = __expf(eh) / (denom[d * 8 + hd] + 1e-8f);
    const float4 hv = *(const float4*)(h + (long)s * 128 + hd * 16 + o4);
    float* op = out + (long)d * 128 + hd * 16 + o4;
    atomicAdd(op + 0, att * hv.x);
    atomicAdd(op + 1, att * hv.y);
    atomicAdd(op + 2, att * hv.z);
    atomicAdd(op + 3, att * hv.w);
}

// ---------------------------------------------------------------------------
// Launcher. Inputs: x, W, a, temporal_weight, edge_index, timestep.
// Workspace: [Wpack bf16 32KB][h f32 N*128][alpha f32 N*16][denom f32 N*8] ~61MB
// ---------------------------------------------------------------------------
extern "C" void kernel_launch(void* const* d_in, const int* in_sizes, int n_in,
                              void* d_out, int out_size, void* d_ws, size_t ws_size,
                              hipStream_t stream) {
    const float* x  = (const float*)d_in[0];
    const float* W  = (const float*)d_in[1];
    const float* a  = (const float*)d_in[2];
    const float* tw = (const float*)d_in[3];
    const int*   ei = (const int*)d_in[4];
    const int*   ts = (const int*)d_in[5];
    float* out = (float*)d_out;

    const int N = in_sizes[5];        // 100000
    const int E = in_sizes[4] / 2;    // 1600000

    char*  ws    = (char*)d_ws;
    __bf16* wpack = (__bf16*)ws;                                   // 32 KB
    float* h     = (float*)(ws + 32768);
    float* alpha = (float*)(ws + 32768 + (size_t)N * 128 * 4);
    float* denom = (float*)(ws + 32768 + (size_t)N * 128 * 4 + (size_t)N * 16 * 4);

    k_pack_W<<<1, 256, 0, stream>>>(W, wpack);

    const int nOut = N * 128;
    k_fill_zero<<<(nOut + 255) / 256, 256, 0, stream>>>(out, nOut);
    k_fill_zero<<<(N * 8 + 255) / 256, 256, 0, stream>>>(denom, N * 8);

    const int waves = (N + 15) / 16;
    k_gemm_wmma<<<(waves + 7) / 8, 256, 0, stream>>>(x, wpack, h, N);

    k_alpha<<<(N * 8 + 255) / 256, 256, 0, stream>>>(h, a, alpha, N);

    k_edge_denom<<<(E + 255) / 256, 256, 0, stream>>>(ei, ts, alpha, tw, denom, E);

    long long t2 = (long long)E * 32;
    k_edge_scatter<<<(int)((t2 + 255) / 256), 256, 0, stream>>>(ei, ts, alpha, tw, denom, h, out, E);
}
